// DepthwiseSeparableFusedConv2d_70720931496354
// MI455X (gfx1250) — 8-bit matrix operand order repaired and checked by reading, compiled, not run on hardware
//
#include <hip/hip_runtime.h>
#include <hip/hip_bf16.h>
#include <stdint.h>


typedef int v8i   __attribute__((ext_vector_type(8)));
typedef unsigned uint32x4 __attribute__((ext_vector_type(4)));
typedef int int32x4 __attribute__((ext_vector_type(4)));
typedef int int32x8 __attribute__((ext_vector_type(8)));

#define BATCH 32
#define CH    256
#define HW    3136
#define BHW   100352
#define NTOT  25690112

__device__ __forceinline__ float mkscale(float mn, float mx) {
    return fmaxf((mx - mn) * (1.0f / 255.0f), 1e-8f);
}
__device__ __forceinline__ float qidx(float v, float mn, float s) {
    return rintf(fminf(fmaxf((v - mn) / s, 0.0f), 255.0f));
}
__device__ __forceinline__ float qzf(float v, float mn, float s) {
    return qidx(v, mn, s) * s + mn;
}

__device__ __forceinline__ void tdm_load_2d(unsigned ldsAddr, const void* gptr,
                                            unsigned rowBytes, unsigned rows,
                                            unsigned strideBytes,
                                            unsigned tdim0, unsigned tdim1) {
    uint64_t ga = (uint64_t)(uintptr_t)gptr;
    uint32x4 g0;
    g0[0] = 1u;
    g0[1] = ldsAddr;
    g0[2] = (unsigned)ga;
    g0[3] = (unsigned)((ga >> 32) & 0x01FFFFFFu) | 0x80000000u;
    int32x8 g1;
    g1[0] = 0;
    g1[1] = (int)((tdim0 & 0xFFFFu) << 16);
    g1[2] = (int)((tdim0 >> 16) | ((tdim1 & 0xFFFFu) << 16));
    g1[3] = (int)((tdim1 >> 16) | (rowBytes << 16));
    g1[4] = (int)rows;
    g1[5] = (int)strideBytes;
    g1[6] = 0;
    g1[7] = 0;
    int32x4 z4 = {0, 0, 0, 0};
#if defined(__clang_major__) && (__clang_major__ >= 23)
    int32x8 z8 = {0, 0, 0, 0, 0, 0, 0, 0};
    __builtin_amdgcn_tensor_load_to_lds(g0, g1, z4, z4, z8, 0);
#else
    __builtin_amdgcn_tensor_load_to_lds(g0, g1, z4, z4, 0);
#endif
}

__device__ __forceinline__ void bredMinMax(float& lo, float& hi, float* smn, float* smx) {
    int t = threadIdx.x;
    smn[t] = lo; smx[t] = hi; __syncthreads();
    for (int s = 128; s > 0; s >>= 1) {
        if (t < s) { smn[t] = fminf(smn[t], smn[t + s]); smx[t] = fmaxf(smx[t], smx[t + s]); }
        __syncthreads();
    }
    lo = smn[0]; hi = smx[0]; __syncthreads();
}


__global__ void __launch_bounds__(256) k_xstats(const float* __restrict__ x, float* __restrict__ xpart) {
    __shared__ float smn[256], smx[256];
    int blk = blockIdx.x, t = threadIdx.x;
    const float* p = x + (long)blk * 12544;
    float lo = 1e30f, hi = -1e30f;
    for (int i = t; i < 12544; i += 256) { float v = p[i]; lo = fminf(lo, v); hi = fmaxf(hi, v); }
    bredMinMax(lo, hi, smn, smx);
    if (t == 0) { xpart[blk * 2] = lo; xpart[blk * 2 + 1] = hi; }
}

__global__ void __launch_bounds__(256) k_params1(const float* __restrict__ dw_w, const float* __restrict__ dw_b,
                                                 const float* __restrict__ xpart, float* __restrict__ Pf,
                                                 int* __restrict__ chunk1) {
    __shared__ float smn[256], smx[256];
    int t = threadIdx.x;
    float lo = 1e30f, hi = -1e30f;
    if (t < 32) {
        for (int j = 0; j < 64; j++) {
            lo = fminf(lo, xpart[(t * 64 + j) * 2]);
            hi = fmaxf(hi, xpart[(t * 64 + j) * 2 + 1]);
        }
    }
    smn[t] = (t < 32) ? lo : 0.0f; smx[t] = (t < 32) ? hi : 0.0f;
    __syncthreads();
    if (t == 0) {
        float a = 0.f, b = 0.f;
        for (int i = 0; i < 32; i++) { a += smn[i]; b += smx[i]; }
        float mn = a / 32.f, mx = b / 32.f;
        Pf[0] = mn; Pf[1] = mkscale(mn, mx);
    }
    __syncthreads();
    lo = 1e30f; hi = -1e30f;
    for (int i = t; i < 2304; i += 256) { float v = dw_w[i]; lo = fminf(lo, v); hi = fmaxf(hi, v); }
    bredMinMax(lo, hi, smn, smx);
    if (t == 0) { Pf[2] = lo; Pf[3] = mkscale(lo, hi); }
    float vb = dw_b[t]; lo = vb; hi = vb;
    bredMinMax(lo, hi, smn, smx);
    if (t == 0) { Pf[4] = lo; Pf[5] = mkscale(lo, hi); }
    for (int i = t; i < 4096; i += 256) {
        chunk1[i] = 2147483647; chunk1[4096 + i] = (-2147483647 - 1); chunk1[8192 + i] = 0;
    }
}

__global__ void __launch_bounds__(256) k_dwconv(const float* __restrict__ x, const float* __restrict__ dw_w,
                                                const float* __restrict__ dw_b, const float* __restrict__ Pf,
                                                float* __restrict__ big, float* __restrict__ pmm) {
    __shared__ float tile[58 * 58];
    __shared__ float smn[256], smx[256];
    int blk = blockIdx.x;
    int b = blk >> 8, c = blk & 255;
    int t = threadIdx.x;
    float mn_x = Pf[0], s_x = Pf[1], mn_w = Pf[2], s_w = Pf[3], mn_b = Pf[4], s_b = Pf[5];
    const float* xp = x + (long)blk * HW;
    for (int l = t; l < 58 * 58; l += 256) {
        int r = l / 58, q = l % 58;
        int ih = r - 1, iw = q - 1;
        float v = 0.f;
        if ((unsigned)ih < 56u && (unsigned)iw < 56u) v = qzf(xp[ih * 56 + iw], mn_x, s_x);
        tile[l] = v;
    }
    float wq[9];
#pragma unroll
    for (int i = 0; i < 9; i++) wq[i] = qzf(dw_w[c * 9 + i], mn_w, s_w);
    float bq = qzf(dw_b[c], mn_b, s_b);
    __syncthreads();
    float lo = 1e30f, hi = -1e30f;
    float* op = big + (long)blk * HW;
    for (int p = t; p < HW; p += 256) {
        int oh = p / 56, ow = p % 56;
        const float* tp = &tile[oh * 58 + ow];
        float acc = bq;
#pragma unroll
        for (int dy = 0; dy < 3; dy++)
#pragma unroll
            for (int dx = 0; dx < 3; dx++)
                acc += tp[dy * 58 + dx] * wq[dy * 3 + dx];
        op[p] = acc;
        lo = fminf(lo, acc); hi = fmaxf(hi, acc);
    }
    bredMinMax(lo, hi, smn, smx);
    if (t == 0) { pmm[blk] = lo; pmm[8192 + blk] = hi; }
    (void)b;
}

__global__ void __launch_bounds__(256) k_params2(const float* __restrict__ pmm, float* __restrict__ Pf) {
    __shared__ float smn[256], smx[256];
    int t = threadIdx.x;
    float am = 0.f, ax = 0.f;
    for (int b = 0; b < 32; b++) {
        float lo = pmm[b * 256 + t], hi = pmm[8192 + b * 256 + t];
        bredMinMax(lo, hi, smn, smx);
        am += lo; ax += hi;
    }
    if (t == 0) { float mn = am / 32.f, mx = ax / 32.f; Pf[6] = mn; Pf[7] = mkscale(mn, mx); }
}

__global__ void __launch_bounds__(256) k_quant_h0(const float* __restrict__ big, const float* __restrict__ Pf,
                                                  unsigned char* __restrict__ q0, int* __restrict__ chunk1,
                                                  int* __restrict__ planeQ) {
    __shared__ int imn[256], imx[256], isum[256];
    int blk = blockIdx.x; int b = blk >> 8, c = blk & 255; int t = threadIdx.x;
    float mn1 = Pf[6], s1 = Pf[7];
    const float* hp = big + (long)blk * HW;
    unsigned char* qp = q0 + (long)blk * HW;
    int lmn = 255, lmx = 0, ls = 0;
    for (int p = t; p < HW; p += 256) {
        int q = (int)qidx(hp[p], mn1, s1);
        qp[p] = (unsigned char)q;
        lmn = min(lmn, q); lmx = max(lmx, q); ls += q;
    }
    imn[t] = lmn; imx[t] = lmx; isum[t] = ls; __syncthreads();
    for (int s = 128; s > 0; s >>= 1) {
        if (t < s) { imn[t] = min(imn[t], imn[t + s]); imx[t] = max(imx[t], imx[t + s]); isum[t] += isum[t + s]; }
        __syncthreads();
    }
    if (t == 0) {
        int chunk = b >> 1;
        atomicMin(&chunk1[c * 16 + chunk], imn[0]);
        atomicMax(&chunk1[4096 + c * 16 + chunk], imx[0]);
        atomicAdd(&chunk1[8192 + c * 16 + chunk], isum[0]);
        planeQ[blk] = imn[0]; planeQ[8192 + blk] = imx[0];
    }
}

__global__ void __launch_bounds__(256) k_params3(const float* __restrict__ bn1_w, const float* __restrict__ bn1_b,
                                                 const float* __restrict__ pw_w, float* __restrict__ Pf,
                                                 const int* __restrict__ chunk1, const int* __restrict__ planeQ) {
    __shared__ float smn[256], smx[256];
    int t = threadIdx.x;
    float mn1 = Pf[6], s1 = Pf[7];
    float mmin = 0.f, mmax = 0.f; long tot = 0;
    for (int k = 0; k < 16; k++) {
        mmin += mn1 + s1 * (float)chunk1[t * 16 + k];
        mmax += mn1 + s1 * (float)chunk1[4096 + t * 16 + k];
        tot  += (long)chunk1[8192 + t * 16 + k];
    }
    mmin *= (1.f / 16.f); mmax *= (1.f / 16.f);
    float meanc = mn1 + s1 * ((float)tot / (float)BHW);
    float sf = (0.5f * 0.35f) * (1.f + sqrtf(3.14159265358979f * logf(4.f))) / sqrtf(2.f * logf(6272.f));
    float scl = 1.f / ((mmax - mmin) * sf + 1e-5f);
    float lo = scl, hi = scl;
    bredMinMax(lo, hi, smn, smx);
    float qscl = qzf(scl, lo, mkscale(lo, hi));
    float wv = bn1_w[t]; lo = wv; hi = wv;
    bredMinMax(lo, hi, smn, smx);
    float qwv = qzf(wv, lo, mkscale(lo, hi));
    float g = qscl * qwv;
    float alpha = s1 * g;
    float beta  = (mn1 - meanc) * g + bn1_b[t];
    Pf[32 + t]  = alpha;
    Pf[288 + t] = beta;
    __syncthreads();
    float accMn = 0.f, accMx = 0.f;
    for (int b = 0; b < 32; b++) {
        float e1 = alpha * (float)planeQ[b * 256 + t] + beta;
        float e2 = alpha * (float)planeQ[8192 + b * 256 + t] + beta;
        lo = fmaxf(0.f, fminf(e1, e2));
        hi = fmaxf(0.f, fmaxf(e1, e2));
        bredMinMax(lo, hi, smn, smx);
        accMn += lo; accMx += hi;
    }
    if (t == 0) { float mn2 = accMn / 32.f, mx2 = accMx / 32.f; Pf[8] = mn2; Pf[9] = mkscale(mn2, mx2); }
    __syncthreads();
    lo = 1e30f; hi = -1e30f;
    for (int i = t; i < 65536; i += 256) { float v = pw_w[i]; lo = fminf(lo, v); hi = fmaxf(hi, v); }
    bredMinMax(lo, hi, smn, smx);
    if (t == 0) { Pf[10] = lo; Pf[11] = mkscale(lo, hi); }
}

__global__ void __launch_bounds__(256) k_qw(const float* __restrict__ pw_w, const float* __restrict__ Pf,
                                            unsigned char* __restrict__ qw, int* __restrict__ Ssum) {
    __shared__ int ssum[256];
    int o = blockIdx.x, t = threadIdx.x;
    float mnw = Pf[10], sw = Pf[11];
    int q = (int)qidx(pw_w[o * 256 + t], mnw, sw);
    qw[o * 256 + t] = (unsigned char)q;
    ssum[t] = q; __syncthreads();
    for (int s = 128; s > 0; s >>= 1) { if (t < s) ssum[t] += ssum[t + s]; __syncthreads(); }
    if (t == 0) Ssum[o] = ssum[0];
}

__global__ void __launch_bounds__(256) k_buildA(const unsigned char* __restrict__ q0, const float* __restrict__ Pf,
                                                unsigned char* __restrict__ qx2, int* __restrict__ Rsum) {
    __shared__ __align__(16) unsigned char rawT[256 * 64];
    __shared__ __align__(16) unsigned char tileT[64 * 256];
    __shared__ int rpart[64][4];
    int blk = blockIdx.x, t = threadIdx.x;
    int b = blk / 49, hwBase = (blk % 49) * 64;
    if ((t >> 5) == 0) {
        tdm_load_2d((unsigned)(uintptr_t)(void*)rawT,
                    q0 + (long)b * 256 * HW + hwBase,
                     64,  256,  HW,
                     64,  256);
        __builtin_amdgcn_s_wait_tensorcnt(0);
    }
    __syncthreads();
    float mn2 = Pf[8], s2 = Pf[9];
    int hwi = t & 63, cg = t >> 6;
    int rs = 0;
    for (int j = 0; j < 64; j++) {
        int c = cg * 64 + j;
        int q = rawT[c * 64 + hwi];
        float h = fmaxf(Pf[32 + c] * (float)q + Pf[288 + c], 0.f);
        int qa = (int)qidx(h, mn2, s2);
        tileT[hwi * 256 + c] = (unsigned char)qa;
        rs += qa;
    }
    rpart[hwi][cg] = rs;
    __syncthreads();
    if (cg == 0)
        Rsum[blk * 64 + hwi] = rpart[hwi][0] + rpart[hwi][1] + rpart[hwi][2] + rpart[hwi][3];
    const int4* src = (const int4*)tileT;
    int4* dst = (int4*)(qx2 + (long)blk * 64 * 256);
    for (int i = t; i < 1024; i += 256) dst[i] = src[i];
}

__global__ void __launch_bounds__(256) k_gemm(const unsigned char* __restrict__ qx2,
                                              const unsigned char* __restrict__ qw,
                                              const int* __restrict__ Rsum, const int* __restrict__ Ssum,
                                              const float* __restrict__ Pf, float* __restrict__ big) {
    __shared__ __align__(16) unsigned char As[64 * 256];
    __shared__ __align__(16) unsigned char Bs[128 * 256];
    int t = threadIdx.x;
    int mBase = blockIdx.x * 64;
    int nBase = blockIdx.y * 128;
    int wave = t >> 5, lane = t & 31;
    if (wave == 0) {
        tdm_load_2d((unsigned)(uintptr_t)(void*)As, qx2 + (long)mBase * 256,
                    256, 64, 256, 256, 64);
        __builtin_amdgcn_s_wait_tensorcnt(0);
    } else if (wave == 1) {
        tdm_load_2d((unsigned)(uintptr_t)(void*)Bs, qw + (long)nBase * 256,
                    256, 128, 256, 256, 128);
        __builtin_amdgcn_s_wait_tensorcnt(0);
    }
    if (mBase + 64 < BHW)
        __builtin_prefetch(qx2 + (long)(mBase + 64) * 256 + t * 64, 0, 1);
    __syncthreads();

    int half = lane >> 4, lr = lane & 15;
    int mSub = wave & 3;
    int nGrp = wave >> 2;
    const int aoff[8] = {0, 4, 16, 20, 32, 36, 48, 52};
    const int boff[8] = {0, 4, 16, 20, 32, 36, 48, 52};
    v8i acc[4] = {};
    const unsigned char* Arow = &As[(mSub * 16 + lr) * 256];
#pragma unroll
    for (int kk = 0; kk < 4; kk++) {
        v8i a;
#pragma unroll
        for (int p = 0; p < 8; p++)
            a[p] = *(const int*)(Arow + kk * 64 + aoff[p] + half * 8);
#pragma unroll
        for (int nt = 0; nt < 4; nt++) {
            const unsigned char* Brow = &Bs[((nGrp * 4 + nt) * 16 + lr) * 256];
            v8i bf;
#pragma unroll
            for (int p = 0; p < 8; p++)
                bf[p] = *(const int*)(Brow + kk * 64 + boff[p] + half * 8);
            acc[nt] = __builtin_amdgcn_wmma_i32_16x16x64_iu8(false, a, false, bf, acc[nt], false, false);
        }
    }
    float mn2 = Pf[8], s2 = Pf[9], mnw = Pf[10], sw = Pf[11];
    float c0 = s2 * sw, c1 = s2 * mnw, c2 = mn2 * sw, c3 = 256.f * mn2 * mnw;
#pragma unroll
    for (int nt = 0; nt < 4; nt++) {
        int n = nBase + (nGrp * 4 + nt) * 16 + lr;
        float sN = c2 * (float)Ssum[n] + c3;
#pragma unroll
        for (int v = 0; v < 8; v++) {
            int m = mBase + mSub * 16 + v + half * 8;
            float val = c0 * (float)acc[nt][v] + c1 * (float)Rsum[m] + sN;
            big[(long)m * 256 + n] = val;
        }
    }
}

__global__ void __launch_bounds__(256) k_h2stats(const float* __restrict__ big, float* __restrict__ gpart) {
    __shared__ float smn[256], smx[256];
    int blk = blockIdx.x, t = threadIdx.x;
    const float* p = big + (long)blk * 64 * 256;
    float lo = 1e30f, hi = -1e30f;
    for (int r = 0; r < 64; r++) { float v = p[r * 256 + t]; lo = fminf(lo, v); hi = fmaxf(hi, v); }
    bredMinMax(lo, hi, smn, smx);
    if (t == 0) { gpart[blk] = lo; gpart[1568 + blk] = hi; }
}

__global__ void __launch_bounds__(256) k_params4(const float* __restrict__ gpart, float* __restrict__ Pf,
                                                 int* __restrict__ chunk2) {
    __shared__ float smn[256], smx[256];
    int t = threadIdx.x;
    float lo = 1e30f, hi = -1e30f;
    if (t < 32) {
        for (int j = 0; j < 49; j++) {
            lo = fminf(lo, gpart[t * 49 + j]);
            hi = fmaxf(hi, gpart[1568 + t * 49 + j]);
        }
    }
    smn[t] = (t < 32) ? lo : 0.f; smx[t] = (t < 32) ? hi : 0.f;
    __syncthreads();
    if (t == 0) {
        float a = 0.f, b = 0.f;
        for (int i = 0; i < 32; i++) { a += smn[i]; b += smx[i]; }
        float mn = a / 32.f, mx = b / 32.f;
        Pf[12] = mn; Pf[13] = mkscale(mn, mx);
    }
    for (int i = t; i < 4096; i += 256) {
        chunk2[i] = 2147483647; chunk2[4096 + i] = (-2147483647 - 1); chunk2[8192 + i] = 0;
    }
}

__global__ void __launch_bounds__(256) k_bn2chunk(const float* __restrict__ big, const float* __restrict__ Pf,
                                                  int* __restrict__ chunk2) {
    int blk = blockIdx.x, t = threadIdx.x;
    float mn3 = Pf[12], s3 = Pf[13];
    const float* p = big + (long)blk * 64 * 256;
    int lmn = 255, lmx = 0, ls = 0;
    for (int r = 0; r < 64; r++) {
        int q = (int)qidx(p[r * 256 + t], mn3, s3);
        lmn = min(lmn, q); lmx = max(lmx, q); ls += q;
    }
    int chunk = blk / 98;
    atomicMin(&chunk2[t * 16 + chunk], lmn);
    atomicMax(&chunk2[4096 + t * 16 + chunk], lmx);
    atomicAdd(&chunk2[8192 + t * 16 + chunk], ls);
}

__global__ void __launch_bounds__(256) k_params5(const float* __restrict__ bn2_w, const float* __restrict__ bn2_b,
                                                 float* __restrict__ Pf, const int* __restrict__ chunk2) {
    __shared__ float smn[256], smx[256];
    int t = threadIdx.x;
    float mn3 = Pf[12], s3 = Pf[13];
    float mmin = 0.f, mmax = 0.f; long tot = 0;
    for (int k = 0; k < 16; k++) {
        mmin += mn3 + s3 * (float)chunk2[t * 16 + k];
        mmax += mn3 + s3 * (float)chunk2[4096 + t * 16 + k];
        tot  += (long)chunk2[8192 + t * 16 + k];
    }
    mmin *= (1.f / 16.f); mmax *= (1.f / 16.f);
    float meanc = mn3 + s3 * ((float)tot / (float)BHW);
    float sf = (0.5f * 0.35f) * (1.f + sqrtf(3.14159265358979f * logf(4.f))) / sqrtf(2.f * logf(6272.f));
    float scl = 1.f / ((mmax - mmin) * sf + 1e-5f);
    float lo = scl, hi = scl;
    bredMinMax(lo, hi, smn, smx);
    float qscl = qzf(scl, lo, mkscale(lo, hi));
    float wv = bn2_w[t]; lo = wv; hi = wv;
    bredMinMax(lo, hi, smn, smx);
    float qwv = qzf(wv, lo, mkscale(lo, hi));
    float g = qscl * qwv;
    Pf[544 + t] = s3 * g;
    Pf[800 + t] = (mn3 - meanc) * g + bn2_b[t];
}

__global__ void __launch_bounds__(256) k_final(const float* __restrict__ big, const float* __restrict__ Pf,
                                               float* __restrict__ out) {
    __shared__ float tile[64 * 65];
    int t = threadIdx.x;
    int mBase = blockIdx.x * 64, oBase = blockIdx.y * 64;
    int b = mBase / HW, hwBase = mBase % HW;
    float mn3 = Pf[12], s3 = Pf[13];
    for (int j = 0; j < 16; j++) {
        int idx = t + j * 256;
        int r = idx >> 6, cc = idx & 63;
        int o = oBase + cc;
        float v = big[(long)(mBase + r) * 256 + o];
        float q = qidx(v, mn3, s3);
        tile[r * 65 + cc] = fmaxf(Pf[544 + o] * q + Pf[800 + o], 0.f);
    }
    __syncthreads();
    for (int j = 0; j < 16; j++) {
        int idx = t + j * 256;
        int oc = idx >> 6, hw = idx & 63;
        out[(long)b * (CH * HW) + (long)(oBase + oc) * HW + hwBase + hw] = tile[hw * 65 + oc];
    }
}

extern "C" void kernel_launch(void* const* d_in, const int* in_sizes, int n_in,
                              void* d_out, int out_size, void* d_ws, size_t ws_size,
                              hipStream_t stream) {
    const float* x     = (const float*)d_in[0];
    const float* dw_w  = (const float*)d_in[1];
    const float* dw_b  = (const float*)d_in[2];
    const float* bn1_w = (const float*)d_in[3];
    const float* bn1_b = (const float*)d_in[4];
    const float* pw_w  = (const float*)d_in[5];
    const float* bn2_w = (const float*)d_in[6];
    const float* bn2_b = (const float*)d_in[7];
    float* out = (float*)d_out;

    char* w = (char*)d_ws;
    float* Pf    = (float*)w; w += 16384;
    float* xpart = (float*)w; w += 2048 * 2 * 4;
    float* pmm   = (float*)w; w += 8192 * 2 * 4;
    int*   planeQ= (int*)w;   w += 8192 * 2 * 4;
    int*   chunk1= (int*)w;   w += 4096 * 3 * 4;
    int*   chunk2= (int*)w;   w += 4096 * 3 * 4;
    int*   Ssum  = (int*)w;   w += 1024;
    int*   Rsum  = (int*)w;   w += BHW * 4;
    float* gpart = (float*)w; w += 1568 * 2 * 4;
    unsigned char* qw  = (unsigned char*)w; w += 65536;
    unsigned char* q0  = (unsigned char*)w; w += NTOT;
    unsigned char* qx2 = (unsigned char*)w; w += NTOT;
    float* big = (float*)w;

    k_xstats  <<<2048, 256, 0, stream>>>(x, xpart);
    k_params1 <<<1,    256, 0, stream>>>(dw_w, dw_b, xpart, Pf, chunk1);
    k_dwconv  <<<8192, 256, 0, stream>>>(x, dw_w, dw_b, Pf, big, pmm);
    k_params2 <<<1,    256, 0, stream>>>(pmm, Pf);
    k_quant_h0<<<8192, 256, 0, stream>>>(big, Pf, q0, chunk1, planeQ);
    k_params3 <<<1,    256, 0, stream>>>(bn1_w, bn1_b, pw_w, Pf, chunk1, planeQ);
    k_qw      <<<256,  256, 0, stream>>>(pw_w, Pf, qw, Ssum);
    k_buildA  <<<1568, 256, 0, stream>>>(q0, Pf, qx2, Rsum);
    k_gemm    <<<dim3(1568, 2), 256, 0, stream>>>(qx2, qw, Rsum, Ssum, Pf, big);
    k_h2stats <<<1568, 256, 0, stream>>>(big, gpart);
    k_params4 <<<1,    256, 0, stream>>>(gpart, Pf, chunk2);
    k_bn2chunk<<<1568, 256, 0, stream>>>(big, Pf, chunk2);
    k_params5 <<<1,    256, 0, stream>>>(bn2_w, bn2_b, Pf, chunk2);
    k_final   <<<dim3(1568, 4), 256, 0, stream>>>(big, Pf, out);
    (void)in_sizes; (void)n_in; (void)out_size; (void)ws_size;
}
